// CrystalGraphConvNet_9216999817911
// MI455X (gfx1250) — matrix operand order repaired and checked by reading, not compiled, not run on hardware
//
#include <hip/hip_runtime.h>
#include <hip/hip_bf16.h>

#define N_ATOMS 50000
#define M_NBR   12
#define F_DIM   256
#define NBR_DIM 41
#define ORIG_DIM 92
#define H_DIM   256
#define N_CONV  3
#define N_CRYS  1000
#define NROWS   (N_ATOMS * M_NBR)   /* 600000 */
#define TWOF    512
#define K_CONV  576                 /* 553 padded to 18*32 */
#define K_EMB   96                  /* 92 padded to 3*32  */

typedef __bf16 bf16;
typedef __attribute__((ext_vector_type(16))) __bf16 v16bf;
typedef __attribute__((ext_vector_type(8)))  __bf16 v8bf;
typedef __attribute__((ext_vector_type(8)))  float  v8f;
typedef __attribute__((ext_vector_type(4)))  int    v4i;

#define AS1 __attribute__((address_space(1)))
#define AS3 __attribute__((address_space(3)))

#if __has_builtin(__builtin_amdgcn_global_load_async_to_lds_b128)
#define USE_ASYNC 1
#endif

static __device__ __forceinline__ void wait_async0() {
#if __has_builtin(__builtin_amdgcn_s_wait_asynccnt)
    __builtin_amdgcn_s_wait_asynccnt(0);
#else
    asm volatile("s_wait_asynccnt 0x0" ::: "memory");
#endif
}
static __device__ __forceinline__ void wait_async3() {
#if __has_builtin(__builtin_amdgcn_s_wait_asynccnt)
    __builtin_amdgcn_s_wait_asynccnt(3);
#else
    asm volatile("s_wait_asynccnt 0x3" ::: "memory");
#endif
}

static __device__ __forceinline__ bf16  f2bf(float x) { return (bf16)x; }
static __device__ __forceinline__ float bf2f(bf16 x)  { return (float)x; }
static __device__ __forceinline__ float sp_f(float x) { return x > 20.f ? x : log1pf(expf(x)); }
static __device__ __forceinline__ float sg_f(float x) { return 1.f / (1.f + expf(-x)); }

// ---------------------------------------------------------------- prep kernels
__global__ void prep_atomfea_kernel(const float* __restrict__ a, bf16* __restrict__ o) {
    long i = (long)blockIdx.x * 256 + threadIdx.x;
    if (i >= (long)N_ATOMS * K_EMB) return;
    int n = (int)(i / K_EMB), k = (int)(i % K_EMB);
    o[i] = f2bf(k < ORIG_DIM ? a[(long)n * ORIG_DIM + k] : 0.f);
}

__global__ void prep_embT_kernel(const float* __restrict__ w, bf16* __restrict__ o) {
    int i = blockIdx.x * 256 + threadIdx.x;
    if (i >= F_DIM * K_EMB) return;
    int col = i / K_EMB, k = i % K_EMB;
    o[i] = f2bf(k < ORIG_DIM ? w[(long)k * F_DIM + col] : 0.f);
}

__global__ void prep_wT_kernel(const float* __restrict__ w, bf16* __restrict__ o) {
    long i = (long)blockIdx.x * 256 + threadIdx.x;
    if (i >= (long)N_CONV * TWOF * K_CONV) return;
    int l   = (int)(i / ((long)TWOF * K_CONV));
    int rem = (int)(i % ((long)TWOF * K_CONV));
    int col = rem / K_CONV, k = rem % K_CONV;
    float v = (k < 2 * F_DIM + NBR_DIM) ? w[(long)l * (2 * F_DIM + NBR_DIM) * TWOF + (long)k * TWOF + col] : 0.f;
    o[i] = f2bf(v);
}

__global__ void prep_nbr_kernel(const float* __restrict__ nbr, bf16* __restrict__ o) {
    long i = (long)blockIdx.x * 256 + threadIdx.x;
    if (i >= (long)NROWS * 64) return;
    long r = i >> 6; int k = (int)(i & 63);
    o[i] = f2bf(k < NBR_DIM ? nbr[r * NBR_DIM + k] : 0.f);
}

__global__ void zero_kernel(float* __restrict__ p, long n) {
    long i = (long)blockIdx.x * 256 + threadIdx.x;
    if (i < n) p[i] = 0.f;
}

// ---------------------------------------------------------------- embedding GEMM
__global__ __launch_bounds__(256) void embed_gemm_kernel(
    const bf16* __restrict__ afea, const bf16* __restrict__ wT,
    const float* __restrict__ bias, float* __restrict__ atom_f32, bf16* __restrict__ atom_bf)
{
#ifdef USE_ASYNC
    __shared__ bf16 lA[2][128][48];
    __shared__ bf16 lB[2][64][48];
#else
    __shared__ bf16 lA[1][128][48];
    __shared__ bf16 lB[1][64][48];
#endif
    const int tid = threadIdx.x, wave = tid >> 5, lane = tid & 31;
    const long rowBase = (long)blockIdx.x * 128;
    const int colBase = blockIdx.y * 64;
    const int sub = tid & 3, r0 = tid >> 2, cB = tid >> 2;
    const int fr = (wave << 4) + (lane & 15);
    const int kb2 = (lane >> 4) << 3;
    const int bh = (lane >> 4) << 4;
    v8f acc[4] = {};

    auto a_src = [&](int kbase, int half) -> const bf16* {
        long gr = rowBase + r0 + half * 64; if (gr >= N_ATOMS) gr = N_ATOMS - 1;
        return afea + gr * K_EMB + kbase + sub * 8;
    };

#ifdef USE_ASYNC
    auto issue_tile = [&](int ks, int buf) {
        const int kbase = ks * 32;
#pragma unroll
        for (int half = 0; half < 2; ++half) {
            __builtin_amdgcn_global_load_async_to_lds_b128(
                (AS1 v4i*)a_src(kbase, half),
                (AS3 v4i*)&lA[buf][r0 + half * 64][sub * 8], 0, 0);
        }
        __builtin_amdgcn_global_load_async_to_lds_b128(
            (AS1 v4i*)(wT + (long)(colBase + cB) * K_EMB + kbase + sub * 8),
            (AS3 v4i*)&lB[buf][cB][sub * 8], 0, 0);
    };
    issue_tile(0, 0);
    for (int ks = 0; ks < 3; ++ks) {
        const int cur = ks & 1;
        wait_async0();
        __syncthreads();
        if (ks + 1 < 3) issue_tile(ks + 1, cur ^ 1);
        v8bf a0 = *(const v8bf*)&lA[cur][fr][bh];
        v8bf a1 = *(const v8bf*)&lA[cur][fr][bh + 8];
        v16bf af;
#pragma unroll
        for (int i = 0; i < 8; ++i) { af[i] = a0[i]; af[i + 8] = a1[i]; }
#pragma unroll
        for (int cf = 0; cf < 4; ++cf) {
            v16bf bf = *(const v16bf*)&lB[cur][cf * 16 + (lane & 15)][bh];
            acc[cf] = __builtin_amdgcn_wmma_f32_16x16x32_bf16(false, af, false, bf, (short)0, acc[cf], false, false);
        }
    }
#else
    for (int ks = 0; ks < 3; ++ks) {
        const int kbase = ks * 32;
        __syncthreads();
#pragma unroll
        for (int half = 0; half < 2; ++half)
            *(v8bf*)&lA[0][r0 + half * 64][sub * 8] = *(const v8bf*)a_src(kbase, half);
        *(v8bf*)&lB[0][cB][sub * 8] =
            *(const v8bf*)(wT + (long)(colBase + cB) * K_EMB + kbase + sub * 8);
        __syncthreads();
        v8bf a0 = *(const v8bf*)&lA[0][fr][bh];
        v8bf a1 = *(const v8bf*)&lA[0][fr][bh + 8];
        v16bf af;
#pragma unroll
        for (int i = 0; i < 8; ++i) { af[i] = a0[i]; af[i + 8] = a1[i]; }
#pragma unroll
        for (int cf = 0; cf < 4; ++cf) {
            v16bf bf = *(const v16bf*)&lB[0][cf * 16 + (lane & 15)][bh];
            acc[cf] = __builtin_amdgcn_wmma_f32_16x16x32_bf16(false, af, false, bf, (short)0, acc[cf], false, false);
        }
    }
#endif
    const int ln = lane & 15, rsel = (lane >> 4) << 3;
#pragma unroll
    for (int cf = 0; cf < 4; ++cf) {
        const int col = colBase + cf * 16 + ln;
        const float bv = bias[col];
#pragma unroll
        for (int j = 0; j < 8; ++j) {
            long rowg = rowBase + (wave << 4) + rsel + j;
            if (rowg < N_ATOMS) {
                float v = acc[cf][j] + bv;
                atom_f32[rowg * F_DIM + col] = v;
                atom_bf [rowg * F_DIM + col] = f2bf(v);
            }
        }
    }
}

// ---------------------------------------------------------------- conv GEMM (main)
// gated(600000,512)bf16 = [self|nbr_gather|nbr_fea](600000,576)bf16 @ wT(512,576)^T + b
// 3-stage async global->LDS pipeline (gfx1250) or reg-staged fallback.
__global__ __launch_bounds__(256) void conv_gemm_kernel(
    const bf16* __restrict__ atom_bf, const bf16* __restrict__ nbr_pad,
    const int*  __restrict__ nbr_idx, const bf16* __restrict__ wT,
    const float* __restrict__ bias, bf16* __restrict__ gated,
    float* __restrict__ col_sum, float* __restrict__ col_sq)
{
#ifdef USE_ASYNC
    __shared__ bf16 lA[3][128][48];
    __shared__ bf16 lB[3][64][48];
#else
    __shared__ bf16 lA[1][128][48];
    __shared__ bf16 lB[1][64][48];
#endif
    __shared__ float lcs[64];
    __shared__ float lcq[64];
    const int tid = threadIdx.x, wave = tid >> 5, lane = tid & 31;
    const long rowBase = (long)blockIdx.x * 128;
    const int colBase = blockIdx.y * 64;
    if (tid < 64) { lcs[tid] = 0.f; lcq[tid] = 0.f; }

    const int sub = tid & 3, r0 = tid >> 2, cB = tid >> 2;
    const int fr = (wave << 4) + (lane & 15);
    const int kb2 = (lane >> 4) << 3;
    const int bh = (lane >> 4) << 4;
    v8f acc[4] = {};

    // per-k-step A-source resolver (self rows / gathered neighbor rows / nbr features)
    auto a_src = [&](int kbase, int half) -> const bf16* {
        const int rr = r0 + half * 64;
        long gr = rowBase + rr; if (gr >= NROWS) gr = NROWS - 1;
        if (kbase < 256)      return atom_bf + (gr / M_NBR) * (long)F_DIM + (kbase + sub * 8);
        else if (kbase < 512) return atom_bf + (long)nbr_idx[gr] * F_DIM + (kbase - 256 + sub * 8);
        else                  return nbr_pad + gr * 64 + (kbase - 512 + sub * 8);
    };

#ifdef USE_ASYNC
    auto issue_tile = [&](int ks, int buf) {
        const int kbase = ks * 32;
#pragma unroll
        for (int half = 0; half < 2; ++half) {
            __builtin_amdgcn_global_load_async_to_lds_b128(
                (AS1 v4i*)a_src(kbase, half),
                (AS3 v4i*)&lA[buf][r0 + half * 64][sub * 8], 0, 0);
        }
        __builtin_amdgcn_global_load_async_to_lds_b128(
            (AS1 v4i*)(wT + (long)(colBase + cB) * K_CONV + kbase + sub * 8),
            (AS3 v4i*)&lB[buf][cB][sub * 8], 0, 0);
    };
    issue_tile(0, 0);          // batch k lives in buffer k % 3
    issue_tile(1, 1);
    for (int ks = 0; ks < 18; ++ks) {
        const int cur = ks % 3;
        if (ks < 17) wait_async3();   // oldest batch (ours) done; one younger batch may fly
        else         wait_async0();   // last step: nothing younger outstanding
        __syncthreads();              // all waves' batch `ks` landed; reads of buf (ks+2)%3 done
        if (ks + 2 < 18) issue_tile(ks + 2, (ks + 2) % 3);
        v8bf a0 = *(const v8bf*)&lA[cur][fr][bh];
        v8bf a1 = *(const v8bf*)&lA[cur][fr][bh + 8];
        v16bf af;
#pragma unroll
        for (int i = 0; i < 8; ++i) { af[i] = a0[i]; af[i + 8] = a1[i]; }
#pragma unroll
        for (int cf = 0; cf < 4; ++cf) {
            v16bf bf = *(const v16bf*)&lB[cur][cf * 16 + (lane & 15)][bh];
            acc[cf] = __builtin_amdgcn_wmma_f32_16x16x32_bf16(false, af, false, bf, (short)0, acc[cf], false, false);
        }
    }
#else
    // register-staged software pipeline fallback
    v8bf rA0 = *(const v8bf*)a_src(0, 0);
    v8bf rA1 = *(const v8bf*)a_src(0, 1);
    v8bf rB  = *(const v8bf*)(wT + (long)(colBase + cB) * K_CONV + 0 + sub * 8);
    for (int ks = 0; ks < 18; ++ks) {
        __syncthreads();
        *(v8bf*)&lA[0][r0][sub * 8]      = rA0;
        *(v8bf*)&lA[0][r0 + 64][sub * 8] = rA1;
        *(v8bf*)&lB[0][cB][sub * 8]      = rB;
        if (ks + 1 < 18) {
            const int kn = (ks + 1) * 32;
            rA0 = *(const v8bf*)a_src(kn, 0);
            rA1 = *(const v8bf*)a_src(kn, 1);
            rB  = *(const v8bf*)(wT + (long)(colBase + cB) * K_CONV + kn + sub * 8);
        }
        __syncthreads();
        v8bf a0 = *(const v8bf*)&lA[0][fr][bh];
        v8bf a1 = *(const v8bf*)&lA[0][fr][bh + 8];
        v16bf af;
#pragma unroll
        for (int i = 0; i < 8; ++i) { af[i] = a0[i]; af[i + 8] = a1[i]; }
#pragma unroll
        for (int cf = 0; cf < 4; ++cf) {
            v16bf bf = *(const v16bf*)&lB[0][cf * 16 + (lane & 15)][bh];
            acc[cf] = __builtin_amdgcn_wmma_f32_16x16x32_bf16(false, af, false, bf, (short)0, acc[cf], false, false);
        }
    }
#endif
    // epilogue: bias, bf16 store, BN1 partial stats (valid rows only)
    const int ln = lane & 15, rsel = (lane >> 4) << 3;
#pragma unroll
    for (int cf = 0; cf < 4; ++cf) {
        const int col = colBase + cf * 16 + ln;
        const float bv = bias[col];
        float ps = 0.f, pq = 0.f;
#pragma unroll
        for (int j = 0; j < 8; ++j) {
            long rowg = rowBase + (wave << 4) + rsel + j;
            if (rowg < NROWS) {
                float v = acc[cf][j] + bv;
                gated[rowg * (long)TWOF + col] = f2bf(v);
                ps += v; pq += v * v;
            }
        }
        atomicAdd(&lcs[cf * 16 + ln], ps);
        atomicAdd(&lcq[cf * 16 + ln], pq);
    }
    __syncthreads();
    if (tid < 64) {
        atomicAdd(&col_sum[colBase + tid], lcs[tid]);
        atomicAdd(&col_sq [colBase + tid], lcq[tid]);
    }
}

// ---------------------------------------------------------------- BN finalize
__global__ void bn_finalize_kernel(const float* __restrict__ s, const float* __restrict__ q,
                                   const float* __restrict__ g, const float* __restrict__ b,
                                   float* __restrict__ scale, float* __restrict__ shift,
                                   float inv_count, int n)
{
    int i = blockIdx.x * 256 + threadIdx.x;
    if (i >= n) return;
    float m  = s[i] * inv_count;
    float v  = q[i] * inv_count - m * m;
    float sc = g[i] * rsqrtf(v + 1e-5f);
    scale[i] = sc;
    shift[i] = b[i] - m * sc;
}

// ---------------------------------------------------------------- gated act + sum over M + BN2 partials
__global__ __launch_bounds__(256) void act_reduce_kernel(
    const bf16* __restrict__ gated, const float* __restrict__ scale1, const float* __restrict__ shift1,
    float* __restrict__ summed, float* __restrict__ bn2_sum, float* __restrict__ bn2_sq)
{
    __shared__ float sc[512];
    __shared__ float sh[512];
    const int tid = threadIdx.x;
    sc[tid] = scale1[tid]; sc[tid + 256] = scale1[tid + 256];
    sh[tid] = shift1[tid]; sh[tid + 256] = shift1[tid + 256];
    __syncthreads();
    const int n0 = blockIdx.x * 8;
    float s2 = 0.f, q2 = 0.f;
    for (int a = 0; a < 8; ++a) {
        const int n = n0 + a;
        if (n >= N_ATOMS) break;
        float s = 0.f;
        for (int m = 0; m < M_NBR; ++m) {
            const long row = (long)(n * M_NBR + m) * TWOF;
            float fv = bf2f(gated[row + tid])       * sc[tid]       + sh[tid];
            float cv = bf2f(gated[row + 256 + tid]) * sc[256 + tid] + sh[256 + tid];
            s += sg_f(fv) * sp_f(cv);
        }
        summed[(long)n * F_DIM + tid] = s;
        s2 += s; q2 += s * s;
    }
    atomicAdd(&bn2_sum[tid], s2);
    atomicAdd(&bn2_sq [tid], q2);
}

// ---------------------------------------------------------------- residual + softplus update
__global__ void update_atom_kernel(const float* __restrict__ summed, const float* __restrict__ scale,
                                   const float* __restrict__ shift, float* __restrict__ atom_f32,
                                   bf16* __restrict__ atom_bf)
{
    long i = (long)blockIdx.x * 256 + threadIdx.x;
    if (i >= (long)N_ATOMS * F_DIM) return;
    int f = (int)(i & 255);
    float v = sp_f(atom_f32[i] + summed[i] * scale[f] + shift[f]);
    atom_f32[i] = v;
    atom_bf [i] = f2bf(v);
}

// ---------------------------------------------------------------- segment pooling
__global__ void pool_kernel(const float* __restrict__ atom, const int* __restrict__ cidx,
                            float* __restrict__ pool, float* __restrict__ cnt)
{
    long i = (long)blockIdx.x * 256 + threadIdx.x;
    if (i >= (long)N_ATOMS * F_DIM) return;
    int n = (int)(i >> 8), f = (int)(i & 255);
    int c = cidx[n];
    atomicAdd(&pool[(long)c * F_DIM + f], atom[i]);
    if (f == 0) atomicAdd(&cnt[c], 1.f);
}

// ---------------------------------------------------------------- head
__global__ __launch_bounds__(256) void head_kernel(
    const float* __restrict__ pool, const float* __restrict__ cnt,
    const float* __restrict__ fc1_w, const float* __restrict__ fc1_b,
    const float* __restrict__ out_w, const float* __restrict__ out_b,
    float* __restrict__ out)
{
    __shared__ float xs[256];
    __shared__ float red[256];
    const int c = blockIdx.x, tid = threadIdx.x;
    float cc = cnt[c]; cc = cc > 1.f ? cc : 1.f;
    xs[tid] = sp_f(pool[(long)c * F_DIM + tid] / cc);
    __syncthreads();
    float a = fc1_b[tid];
    for (int k = 0; k < F_DIM; ++k) a += xs[k] * fc1_w[(long)k * H_DIM + tid];
    red[tid] = sp_f(a) * out_w[tid];
    __syncthreads();
    for (int s = 128; s > 0; s >>= 1) {
        if (tid < s) red[tid] += red[tid + s];
        __syncthreads();
    }
    if (tid == 0) out[c] = red[0] + out_b[0];
}

// ---------------------------------------------------------------- launch
extern "C" void kernel_launch(void* const* d_in, const int* in_sizes, int n_in,
                              void* d_out, int out_size, void* d_ws, size_t ws_size,
                              hipStream_t stream) {
    (void)in_sizes; (void)n_in; (void)out_size; (void)ws_size;
    const float* atom_fea  = (const float*)d_in[0];
    const float* nbr_fea   = (const float*)d_in[1];
    const int*   nbr_idx   = (const int*)  d_in[2];
    const int*   cidx      = (const int*)  d_in[3];
    const float* emb_w     = (const float*)d_in[4];
    const float* emb_b     = (const float*)d_in[5];
    const float* fc_full_w = (const float*)d_in[6];
    const float* fc_full_b = (const float*)d_in[7];
    const float* bn1_g     = (const float*)d_in[8];
    const float* bn1_b     = (const float*)d_in[9];
    const float* bn2_g     = (const float*)d_in[10];
    const float* bn2_b     = (const float*)d_in[11];
    const float* fc1_w     = (const float*)d_in[12];
    const float* fc1_b     = (const float*)d_in[13];
    const float* out_w     = (const float*)d_in[14];
    const float* out_b     = (const float*)d_in[15];
    float* out = (float*)d_out;

    char* ws = (char*)d_ws;
    size_t off = 0;
    auto take = [&](size_t bytes) -> char* {
        char* p = ws + off;
        off = (off + bytes + 255) & ~(size_t)255;
        return p;
    };
    float* atom_f32 = (float*)take((size_t)N_ATOMS * F_DIM * 4);
    bf16*  atom_bf  = (bf16*) take((size_t)N_ATOMS * F_DIM * 2);
    bf16*  afea_bf  = (bf16*) take((size_t)N_ATOMS * K_EMB * 2);
    bf16*  nbr_pad  = (bf16*) take((size_t)NROWS * 64 * 2);
    bf16*  wT       = (bf16*) take((size_t)N_CONV * TWOF * K_CONV * 2);
    bf16*  embT     = (bf16*) take((size_t)F_DIM * K_EMB * 2);
    bf16*  gated    = (bf16*) take((size_t)NROWS * TWOF * 2);
    float* summed   = (float*)take((size_t)N_ATOMS * F_DIM * 4);
    float* stats    = (float*)take(1536 * 4);
    float* bn1_scale= (float*)take(512 * 4);
    float* bn1_shift= (float*)take(512 * 4);
    float* bn2_scale= (float*)take(256 * 4);
    float* bn2_shift= (float*)take(256 * 4);
    float* poolcnt  = (float*)take((size_t)(N_CRYS * F_DIM + N_CRYS) * 4);
    float* bn1_sum = stats;        float* bn1_sq = stats + 512;
    float* bn2_sum = stats + 1024; float* bn2_sq = stats + 1280;
    float* pool = poolcnt;         float* cnt = poolcnt + (size_t)N_CRYS * F_DIM;

    dim3 blk(256);
    prep_atomfea_kernel<<<(N_ATOMS * K_EMB + 255) / 256, blk, 0, stream>>>(atom_fea, afea_bf);
    prep_embT_kernel<<<(F_DIM * K_EMB + 255) / 256, blk, 0, stream>>>(emb_w, embT);
    prep_wT_kernel<<<(unsigned)(((long)N_CONV * TWOF * K_CONV + 255) / 256), blk, 0, stream>>>(fc_full_w, wT);
    prep_nbr_kernel<<<(unsigned)(((long)NROWS * 64 + 255) / 256), blk, 0, stream>>>(nbr_fea, nbr_pad);

    embed_gemm_kernel<<<dim3((N_ATOMS + 127) / 128, F_DIM / 64), blk, 0, stream>>>(
        afea_bf, embT, emb_b, atom_f32, atom_bf);

    for (int l = 0; l < N_CONV; ++l) {
        zero_kernel<<<(1536 + 255) / 256, blk, 0, stream>>>(stats, 1536);
        conv_gemm_kernel<<<dim3((NROWS + 127) / 128, TWOF / 64), blk, 0, stream>>>(
            atom_bf, nbr_pad, nbr_idx, wT + (size_t)l * TWOF * K_CONV,
            fc_full_b + (size_t)l * TWOF, gated, bn1_sum, bn1_sq);
        bn_finalize_kernel<<<2, blk, 0, stream>>>(bn1_sum, bn1_sq,
            bn1_g + (size_t)l * TWOF, bn1_b + (size_t)l * TWOF,
            bn1_scale, bn1_shift, 1.f / (float)NROWS, TWOF);
        act_reduce_kernel<<<(N_ATOMS + 7) / 8, blk, 0, stream>>>(
            gated, bn1_scale, bn1_shift, summed, bn2_sum, bn2_sq);
        bn_finalize_kernel<<<1, blk, 0, stream>>>(bn2_sum, bn2_sq,
            bn2_g + (size_t)l * F_DIM, bn2_b + (size_t)l * F_DIM,
            bn2_scale, bn2_shift, 1.f / (float)N_ATOMS, F_DIM);
        update_atom_kernel<<<(unsigned)(((long)N_ATOMS * F_DIM + 255) / 256), blk, 0, stream>>>(
            summed, bn2_scale, bn2_shift, atom_f32, atom_bf);
    }

    zero_kernel<<<(N_CRYS * F_DIM + N_CRYS + 255) / 256, blk, 0, stream>>>(
        poolcnt, (long)N_CRYS * F_DIM + N_CRYS);
    pool_kernel<<<(unsigned)(((long)N_ATOMS * F_DIM + 255) / 256), blk, 0, stream>>>(
        atom_f32, cidx, pool, cnt);
    head_kernel<<<N_CRYS, blk, 0, stream>>>(pool, cnt, fc1_w, fc1_b, out_w, out_b, out);
}
